// Attention_51591147159669
// MI455X (gfx1250) — compile-verified, matrix operand order checked by reading, not run on hardware
//
#include <hip/hip_runtime.h>

// ---------------------------------------------------------------------------
// Attention decoder (attention-OCR style) for MI455X / gfx1250, wave32 + WMMA.
// GEMMs: v_wmma_f32_16x16x32_f16 with 2x4 register blocking (32x64 per wave).
// batch_H / feat / weights converted to f16 once so the per-step attention
// streams (feat 67MB + batch_H 67MB) stay resident in the 192MB L2.
// Attention score phase double-buffers feature rows through LDS with
// global_load_async_to_lds_b128 / s_wait_asynccnt (guarded, with fallback).
// ---------------------------------------------------------------------------

typedef __attribute__((ext_vector_type(16))) _Float16 v16h;
typedef __attribute__((ext_vector_type(8)))  _Float16 v8h;
typedef __attribute__((ext_vector_type(2)))  _Float16 v2h;
typedef __attribute__((ext_vector_type(8)))  float    v8f;
typedef __attribute__((ext_vector_type(4)))  int      v4i;

constexpr int cB = 512, cT = 128, cI = 512, cH = 512, cC = 97, cS = 26;
constexpr int cX  = 1152;      // padded concat width: [ctx(512) | onehot(97) | h(512) | pad(31)]
constexpr int cG  = 4 * cH;    // 2048 LSTM gates
constexpr int cNG = 128;       // generator rows padded to 8 N-tiles

#if defined(__has_builtin)
#if __has_builtin(__builtin_amdgcn_global_load_async_to_lds_b128)
#define ATT_ASYNC_LDS 1
#endif
#endif
#ifndef ATT_ASYNC_LDS
#define ATT_ASYNC_LDS 0
#endif

#if ATT_ASYNC_LDS
// builtin signature (from the compiler diagnostic): first param is a v4i* in
// the global (AS1) address space, second the LDS (AS3) destination, then the
// immediate instruction offset and cache-policy.
typedef __attribute__((address_space(1))) v4i* gv4ip;
typedef __attribute__((address_space(3))) v4i* lv4ip;
// Copy one 1KB (512 x f16) feature row to LDS: 16B per lane, two async ops
// sharing the per-lane addresses via INST_OFFSET (applied to both sides).
__device__ __forceinline__ void async_copy_row(const _Float16* src, _Float16* dst) {
  __builtin_amdgcn_global_load_async_to_lds_b128((gv4ip)src, (lv4ip)dst, 0,   0);
  __builtin_amdgcn_global_load_async_to_lds_b128((gv4ip)src, (lv4ip)dst, 512, 0);
}
#endif

// ---------------------------------------------------------------------------
// WMMA fragment loader. For 16-bit A (16x32 MxK), lane l in [0,16) holds row
// M=l, lanes [16,32) hold the same rows with K shifted by 8:
//   elements 0..7  -> K = kbase + 0..7      (kbase = 8*(l>=16))
//   elements 8..15 -> K = kbase + 16..23
// The B (32x16 KxN) fragment is exactly the A layout applied to Bw[N,K].
// ---------------------------------------------------------------------------
__device__ __forceinline__ v16h load_frag(const _Float16* __restrict__ p, int ld,
                                          int r0, int k0, int lane) {
  const _Float16* q = p + (size_t)(r0 + (lane & 15)) * ld + k0 + ((lane >> 4) << 3);
  v8h lo = *(const v8h*)q;
  v8h hi = *(const v8h*)(q + 16);
  v16h r;
#pragma unroll
  for (int i = 0; i < 8; ++i) { r[i] = lo[i]; r[i + 8] = hi[i]; }
  return r;
}

__device__ __forceinline__ v8f wmma_f16(v16h a, v16h b, v8f c) {
  return __builtin_amdgcn_wmma_f32_16x16x32_f16(false, a, false, b,
                                                (short)0, c, false, false);
}

// Generic NT GEMM: C[M,N] = A[M,K] * Bw[N,K]^T + bias[N].
// Each wave computes a 32x64 macro-tile (2x4 WMMA tiles, 8 f32 accumulators):
// 6 fragment loads feed 8 WMMAs (0.75 loads/WMMA).
// Mtiles must be even, Ntiles a multiple of 4.
template<bool OUT_F16, bool GUARD_N>
__global__ __launch_bounds__(256) void wmma_gemm_nt(
    const _Float16* __restrict__ A, int lda,
    const _Float16* __restrict__ Bw, int ldb,
    const float* __restrict__ bias,
    void* __restrict__ Cout, int ldc,
    int Mtiles, int Ntiles, int Ktiles, int Nreal)
{
  int wid  = (blockIdx.x * blockDim.x + threadIdx.x) >> 5;
  int lane = threadIdx.x & 31;
  int Ms = Mtiles >> 1, Ns = Ntiles >> 2;
  if (wid >= Ms * Ns) return;               // wave-uniform: EXEC stays all-ones
  int sm = wid / Ns, sn = wid - sm * Ns;
  int r0 = sm << 5, c0 = sn << 6;
  v8f acc[2][4] = {};
  for (int kt = 0; kt < Ktiles; ++kt) {
    int k0 = kt << 5;
    // speculative gfx1250 prefetch of the next K chunk (harmless past the end)
    __builtin_prefetch(A  + (size_t)(r0 + (lane & 15)) * lda + k0 + 32, 0, 3);
    __builtin_prefetch(Bw + (size_t)(c0 + (lane & 15)) * ldb + k0 + 32, 0, 3);
    v16h a[2], b[4];
#pragma unroll
    for (int i = 0; i < 2; ++i) a[i] = load_frag(A,  lda, r0 + 16 * i, k0, lane);
#pragma unroll
    for (int j = 0; j < 4; ++j) b[j] = load_frag(Bw, ldb, c0 + 16 * j, k0, lane);
#pragma unroll
    for (int i = 0; i < 2; ++i)
#pragma unroll
      for (int j = 0; j < 4; ++j)
        acc[i][j] = wmma_f16(a[i], b[j], acc[i][j]);
  }
#pragma unroll
  for (int j = 0; j < 4; ++j) {
    int n = c0 + 16 * j + (lane & 15);
    bool ok = !GUARD_N || (n < Nreal);
    float bv = (bias != nullptr && ok) ? bias[n] : 0.0f;
    if (ok) {
#pragma unroll
      for (int i = 0; i < 2; ++i) {
#pragma unroll
        for (int r = 0; r < 8; ++r) {       // C/D layout: m = r + 8*(lane>=16)
          int m = r0 + 16 * i + r + ((lane >> 4) << 3);
          float v = acc[i][j][r] + bv;
          if (OUT_F16) ((_Float16*)Cout)[(size_t)m * ldc + n] = (_Float16)v;
          else         ((float*)    Cout)[(size_t)m * ldc + n] = v;
        }
      }
    }
  }
}

// ---------------------------------------------------------------------------
// Attention: e_t = sum_h tanh(feat[b,t,h] + hp[b,h]) * w[h]; softmax over T;
// ctx[b,i] = sum_t alpha_t * H16[b,t,i], written as f16 into xbuf[:, 0:512].
// One block (8 waves) per batch row. Lane l owns the 16 contiguous h-values
// [16l, 16l+16); hp/w slices hoisted to registers. Score phase double-buffers
// feature rows through LDS via async loads when available.
// ---------------------------------------------------------------------------
__global__ __launch_bounds__(256) void attention_ctx(
    const _Float16* __restrict__ feat, const float* __restrict__ hp,
    const float* __restrict__ w_score, const _Float16* __restrict__ H16,
    _Float16* __restrict__ xbuf)
{
  __shared__ float sE[cT];
  __shared__ float sRed[cT];
#if ATT_ASYNC_LDS
  __shared__ _Float16 sF[2][8][cH];   // 2 buffers x 8 waves x 512 halves = 16KB
#endif
  int b = blockIdx.x;
  int lane = threadIdx.x & 31, wv = threadIdx.x >> 5, nw = blockDim.x >> 5;

  // t-invariant per-lane slices of hp and w_score (h = 16*lane + j)
  float hpv[16], wsv[16];
  {
    const float4* hp4 = (const float4*)(hp + (size_t)b * cH) + lane * 4;
    const float4* w4  = (const float4*)w_score + lane * 4;
#pragma unroll
    for (int q = 0; q < 4; ++q) {
      float4 a = hp4[q], c = w4[q];
      hpv[4 * q + 0] = a.x; hpv[4 * q + 1] = a.y; hpv[4 * q + 2] = a.z; hpv[4 * q + 3] = a.w;
      wsv[4 * q + 0] = c.x; wsv[4 * q + 1] = c.y; wsv[4 * q + 2] = c.z; wsv[4 * q + 3] = c.w;
    }
  }

  const _Float16* fb = feat + (size_t)b * cT * cH;
#if ATT_ASYNC_LDS
  // software pipeline: wave wv handles rows t = wv, wv+8, ...; copy row t+8
  // into the other LDS buffer while computing row t from the current one.
  async_copy_row(fb + (size_t)wv * cH + lane * 8, &sF[0][wv][lane * 8]);
  int buf = 0;
  for (int t = wv; t < cT; t += nw, buf ^= 1) {
    int tnext = t + nw;
    if (tnext < cT) {
      async_copy_row(fb + (size_t)tnext * cH + lane * 8, &sF[buf ^ 1][wv][lane * 8]);
      asm volatile("s_wait_asynccnt 0x2" ::: "memory");   // current buffer landed
    } else {
      asm volatile("s_wait_asynccnt 0x0" ::: "memory");
    }
    const v8h* row = (const v8h*)(&sF[buf][wv][0]);
    v8h f0 = row[lane * 2], f1 = row[lane * 2 + 1];
    float s = 0.0f;
#pragma unroll
    for (int j = 0; j < 8; ++j) s += tanhf((float)f0[j] + hpv[j]) * wsv[j];
#pragma unroll
    for (int j = 0; j < 8; ++j) s += tanhf((float)f1[j] + hpv[8 + j]) * wsv[8 + j];
#pragma unroll
    for (int off = 16; off > 0; off >>= 1) s += __shfl_xor(s, off, 32);
    if (lane == 0) sE[t] = s;
  }
#else
  for (int t = wv; t < cT; t += nw) {
    const v8h* fr = (const v8h*)(fb + (size_t)t * cH);
    if (t + nw < cT)
      __builtin_prefetch(fb + (size_t)(t + nw) * cH + lane * 16, 0, 3);
    v8h f0 = fr[lane * 2], f1 = fr[lane * 2 + 1];
    float s = 0.0f;
#pragma unroll
    for (int j = 0; j < 8; ++j) s += tanhf((float)f0[j] + hpv[j]) * wsv[j];
#pragma unroll
    for (int j = 0; j < 8; ++j) s += tanhf((float)f1[j] + hpv[8 + j]) * wsv[8 + j];
#pragma unroll
    for (int off = 16; off > 0; off >>= 1) s += __shfl_xor(s, off, 32);
    if (lane == 0) sE[t] = s;
  }
#endif
  __syncthreads();

  // softmax over T=128 in LDS
  if (threadIdx.x < cT) sRed[threadIdx.x] = sE[threadIdx.x];
  __syncthreads();
  for (int off = cT / 2; off > 0; off >>= 1) {
    if (threadIdx.x < off)
      sRed[threadIdx.x] = fmaxf(sRed[threadIdx.x], sRed[threadIdx.x + off]);
    __syncthreads();
  }
  float mx = sRed[0];
  __syncthreads();
  if (threadIdx.x < cT) {
    float e = __expf(sE[threadIdx.x] - mx);
    sE[threadIdx.x] = e;
    sRed[threadIdx.x] = e;
  }
  __syncthreads();
  for (int off = cT / 2; off > 0; off >>= 1) {
    if (threadIdx.x < off) sRed[threadIdx.x] += sRed[threadIdx.x + off];
    __syncthreads();
  }
  float inv = 1.0f / sRed[0];
  __syncthreads();
  if (threadIdx.x < cT) sE[threadIdx.x] *= inv;
  __syncthreads();

  // context: thread tid owns columns {2*tid, 2*tid+1}; coalesced v2h loads
  const v2h* Hb = (const v2h*)(H16 + (size_t)b * cT * cI);
  int tid = threadIdx.x;
  float a0 = 0.0f, a1 = 0.0f;
  for (int t = 0; t < cT; ++t) {
    v2h hv = Hb[(size_t)t * (cI / 2) + tid];
    float al = sE[t];
    a0 += al * (float)hv[0];
    a1 += al * (float)hv[1];
  }
  v2h o; o[0] = (_Float16)a0; o[1] = (_Float16)a1;
  *(v2h*)(xbuf + (size_t)b * cX + 2 * tid) = o;
}

// onehot + previous-h into the concat buffer (ctx was written by attention_ctx)
__global__ void pack_step(const int* __restrict__ text, int s,
                          const _Float16* __restrict__ hf,
                          _Float16* __restrict__ xbuf)
{
  int b = blockIdx.x;
  _Float16* xb = xbuf + (size_t)b * cX;
  for (int c = threadIdx.x; c < cC; c += blockDim.x) xb[cI + c] = (_Float16)0.0f;
  for (int i = threadIdx.x; i < cH; i += blockDim.x) xb[cI + cC + i] = hf[(size_t)b * cH + i];
  __syncthreads();
  if (threadIdx.x == 0) xb[cI + text[(size_t)b * cS + s]] = (_Float16)1.0f;
}

__device__ __forceinline__ float sigmoidf(float x) { return 1.0f / (1.0f + __expf(-x)); }

__global__ void lstm_step(const float* __restrict__ gates, float* __restrict__ cst,
                          _Float16* __restrict__ hf, _Float16* __restrict__ hseq, int s)
{
  int idx = blockIdx.x * blockDim.x + threadIdx.x;
  if (idx >= cB * cH) return;
  int b = idx >> 9, h = idx & (cH - 1);
  const float* g = gates + (size_t)b * cG;   // PyTorch gate order [i,f,g,o]
  float ig = sigmoidf(g[h]);
  float fg = sigmoidf(g[cH + h]);
  float gg = tanhf(g[2 * cH + h]);
  float og = sigmoidf(g[3 * cH + h]);
  float c  = fg * cst[idx] + ig * gg;
  float hh = og * tanhf(c);
  cst[idx] = c;
  hf[idx]  = (_Float16)hh;
  hseq[((size_t)b * cS + s) * cH + h] = (_Float16)hh;
}

// ---- small conversion / packing helpers -----------------------------------
__global__ void cvt_f16(const float* __restrict__ s, _Float16* __restrict__ d, size_t n) {
  size_t i = (size_t)blockIdx.x * blockDim.x + threadIdx.x;
  if (i < n) d[i] = (_Float16)s[i];
}
__global__ void zero_f16(_Float16* __restrict__ p, size_t n) {
  size_t i = (size_t)blockIdx.x * blockDim.x + threadIdx.x;
  if (i < n) p[i] = (_Float16)0.0f;
}
__global__ void zero_f32(float* __restrict__ p, size_t n) {
  size_t i = (size_t)blockIdx.x * blockDim.x + threadIdx.x;
  if (i < n) p[i] = 0.0f;
}
__global__ void pack_wcat(const float* __restrict__ Wih, const float* __restrict__ Whh,
                          _Float16* __restrict__ Wc) {
  int idx = blockIdx.x * blockDim.x + threadIdx.x;
  if (idx >= cG * cX) return;
  int r = idx / cX, c = idx - r * cX;
  float v;
  if (c < cI + cC)            v = Wih[(size_t)r * (cI + cC) + c];
  else if (c < cI + cC + cH)  v = Whh[(size_t)r * cH + (c - cI - cC)];
  else                        v = 0.0f;
  Wc[idx] = (_Float16)v;
}
__global__ void pack_wgen(const float* __restrict__ Wg, _Float16* __restrict__ Wp) {
  int idx = blockIdx.x * blockDim.x + threadIdx.x;
  if (idx >= cNG * cH) return;
  int r = idx / cH, c = idx - r * cH;
  Wp[idx] = (_Float16)(r < cC ? Wg[(size_t)r * cH + c] : 0.0f);
}
__global__ void add_bias(const float* __restrict__ a, const float* __restrict__ b,
                         float* __restrict__ o, int n) {
  int i = blockIdx.x * blockDim.x + threadIdx.x;
  if (i < n) o[i] = a[i] + b[i];
}

// ---------------------------------------------------------------------------
extern "C" void kernel_launch(void* const* d_in, const int* in_sizes, int n_in,
                              void* d_out, int out_size, void* d_ws, size_t ws_size,
                              hipStream_t stream)
{
  const float* batch_H = (const float*)d_in[0];
  const int*   text    = (const int*)  d_in[1];
  const float* W_feat  = (const float*)d_in[2];
  const float* W_hid   = (const float*)d_in[3];
  const float* b_hid   = (const float*)d_in[4];
  const float* w_score = (const float*)d_in[5];
  const float* W_ih    = (const float*)d_in[6];
  const float* W_hh    = (const float*)d_in[7];
  const float* b_ih    = (const float*)d_in[8];
  const float* b_hh    = (const float*)d_in[9];
  const float* W_gen   = (const float*)d_in[10];
  const float* b_gen   = (const float*)d_in[11];
  float* out = (float*)d_out;

  // workspace carve-out
  char* w = (char*)d_ws;
  size_t off = 0;
  auto take = [&](size_t bytes) {
    void* p = w + off;
    off = (off + bytes + 255) & ~(size_t)255;
    return p;
  };
  _Float16* H16    = (_Float16*)take((size_t)cB * cT * cI * 2);  // batch_H f16
  _Float16* feat16 = (_Float16*)take((size_t)cB * cT * cH * 2);  // projected features f16
  _Float16* Wf16   = (_Float16*)take((size_t)cH * cI * 2);
  _Float16* Whid16 = (_Float16*)take((size_t)cH * cH * 2);
  _Float16* Wcat16 = (_Float16*)take((size_t)cG * cX * 2);
  _Float16* Wgen16 = (_Float16*)take((size_t)cNG * cH * 2);
  float*    biasg  = (float*)   take((size_t)cG * 4);
  _Float16* hf16   = (_Float16*)take((size_t)cB * cH * 2);
  float*    cstate = (float*)   take((size_t)cB * cH * 4);
  float*    hpbuf  = (float*)   take((size_t)cB * cH * 4);
  _Float16* xbuf   = (_Float16*)take((size_t)cB * cX * 2);
  float*    gates  = (float*)   take((size_t)cB * cG * 4);
  _Float16* hseq   = (_Float16*)take((size_t)cB * cS * cH * 2);

  auto g1 = [](size_t n) { return dim3((unsigned)((n + 255) / 256)); };
  // waves = (Mtiles/2)*(Ntiles/4), 8 waves per 256-thread block
  auto gw = [](int Mt, int Nt) {
    size_t waves = (size_t)(Mt >> 1) * (size_t)(Nt >> 2);
    return dim3((unsigned)((waves * 32 + 255) / 256));
  };

  // one-time conversions / packing
  cvt_f16 <<<g1((size_t)cB * cT * cI), 256, 0, stream>>>(batch_H, H16, (size_t)cB * cT * cI);
  cvt_f16 <<<g1((size_t)cH * cI),      256, 0, stream>>>(W_feat,  Wf16, (size_t)cH * cI);
  cvt_f16 <<<g1((size_t)cH * cH),      256, 0, stream>>>(W_hid, Whid16, (size_t)cH * cH);
  pack_wcat<<<g1((size_t)cG * cX),     256, 0, stream>>>(W_ih, W_hh, Wcat16);
  pack_wgen<<<g1((size_t)cNG * cH),    256, 0, stream>>>(W_gen, Wgen16);
  add_bias <<<g1(cG),                  256, 0, stream>>>(b_ih, b_hh, biasg, cG);
  zero_f16 <<<g1((size_t)cB * cH),     256, 0, stream>>>(hf16,  (size_t)cB * cH);
  zero_f32 <<<g1((size_t)cB * cH),     256, 0, stream>>>(cstate,(size_t)cB * cH);
  zero_f16 <<<g1((size_t)cB * cX),     256, 0, stream>>>(xbuf,  (size_t)cB * cX);  // also pads cols 1121..1151

  // feat = batch_H @ W_feat^T : [B*T, H], f16 output
  {
    int Mt = (cB * cT) / 16, Nt = cH / 16, Kt = cI / 32;
    wmma_gemm_nt<true,  false><<<gw(Mt, Nt), 256, 0, stream>>>(
        H16, cI, Wf16, cI, nullptr, feat16, cH, Mt, Nt, Kt, cH);
  }

  for (int s = 0; s < cS; ++s) {
    // hp = h @ W_hid^T + b_hid
    {
      int Mt = cB / 16, Nt = cH / 16, Kt = cH / 32;
      wmma_gemm_nt<false, false><<<gw(Mt, Nt), 256, 0, stream>>>(
          hf16, cH, Whid16, cH, b_hid, hpbuf, cH, Mt, Nt, Kt, cH);
    }
    // scores + softmax + context -> xbuf[:, 0:512]
    attention_ctx<<<cB, 256, 0, stream>>>(feat16, hpbuf, w_score, H16, xbuf);
    // onehot + previous h -> xbuf[:, 512:1121]
    pack_step<<<cB, 256, 0, stream>>>(text, s, hf16, xbuf);
    // gates = [ctx|oh|h] @ [W_ih|W_hh]^T + (b_ih + b_hh)
    {
      int Mt = cB / 16, Nt = cG / 16, Kt = cX / 32;
      wmma_gemm_nt<false, false><<<gw(Mt, Nt), 256, 0, stream>>>(
          xbuf, cX, Wcat16, cX, biasg, gates, cG, Mt, Nt, Kt, cG);
    }
    // pointwise LSTM update
    lstm_step<<<g1((size_t)cB * cH), 256, 0, stream>>>(gates, cstate, hf16, hseq, s);
  }

  // probs = out_h @ W_gen^T + b_gen : [B*S, 97] (N padded to 128, guarded store)
  {
    int Mt = (cB * cS) / 16, Nt = cNG / 16, Kt = cH / 32;
    wmma_gemm_nt<false, true><<<gw(Mt, Nt), 256, 0, stream>>>(
        hseq, cH, Wgen16, cH, b_gen, out, cC, Mt, Nt, Kt, cC);
  }
  (void)in_sizes; (void)n_in; (void)out_size; (void)ws_size;
}
